// RecurrentGeneralizedPFAttention_83519934038334
// MI455X (gfx1250) — matrix operand order repaired and checked by reading, not compiled, not run on hardware
//
#include <hip/hip_runtime.h>
#include <hip/hip_bf16.h>

// CDNA5 / gfx1250: wave32, WMMA bf16 16x16x32 with f32 accumulate.
typedef __attribute__((ext_vector_type(16))) __bf16 v16bf;
typedef __attribute__((ext_vector_type(8)))  float  v8f;

constexpr int Bc = 4, Nc = 1024, Dc = 512, Hc = 8, DHc = 64;
constexpr int HDc  = Hc * DHc;        // 512
constexpr int QKVc = 3 * HDc;         // 1536
constexpr float SCALEc = 0.125f;      // 64^-0.5
constexpr int CHUNK = 128;            // score row chunk (QR chunk stays L2-resident)
constexpr int QR_ROWS = CHUNK + 16;   // 144 (tile-padded, +1 row needed by rel-shift)

__device__ inline void cvt4(v16bf& v, int base, float4 f) {
  v[base + 0] = (__bf16)f.x; v[base + 1] = (__bf16)f.y;
  v[base + 2] = (__bf16)f.z; v[base + 3] = (__bf16)f.w;
}

// ---- WMMA fragment loaders (f32 source -> bf16 fragments, b128 loads) ------
// A: [M,K] row-major. Lane L holds row (L&15); elems 0..7 -> K = k0+(L>>4)*8+e,
// elems 8..15 -> K = k0+16+(L>>4)*8+(e-8).  (ISA 7.12.2, 16-bit A 16x32)
__device__ inline v16bf frag_a_f32(const float* __restrict__ A, int lda, int row, int k0, int lane) {
  const float* p0 = A + (size_t)(row + (lane & 15)) * lda + k0 + (lane >> 4) * 8;
  const float4* q0 = (const float4*)p0;        // 32B-aligned by construction
  const float4* q1 = (const float4*)(p0 + 16);
  v16bf a;
  cvt4(a, 0, q0[0]); cvt4(a, 4, q0[1]);
  cvt4(a, 8, q1[0]); cvt4(a, 12, q1[1]);
  return a;
}

// B: from BT [N,K] row-major (transposed-B GEMM). Lane L holds col (L&15);
// elems 0..7 -> K = k0+(L>>4)*8+e, elems 8..15 -> K = k0+16+(L>>4)*8+(e-8).
__device__ inline v16bf frag_b_f32(const float* __restrict__ BT, int ldb, int col, int k0, int lane) {
  const float* p = BT + (size_t)(col + (lane & 15)) * ldb + k0 + (lane >> 4) * 8;
  const float4* q = (const float4*)p;          // 32B-aligned by construction
  const float4* q1 = (const float4*)(p + 16);
  v16bf b;
  cvt4(b, 0, q[0]); cvt4(b, 4, q[1]); cvt4(b, 8, q1[0]); cvt4(b, 12, q1[1]);
  return b;
}

#define WMMA_BF16(acc, a, b) \
  __builtin_amdgcn_wmma_f32_16x16x32_bf16(false, (a), false, (b), (short)0, (acc), false, false)

// C/D layout: lane L, VGPR r -> row = row0 + r + 8*(L>>4), col = col0 + (L&15).

// ---- register-blocked GEMM core: MTx NT tiles of 16x16 per wave ------------
template <int MT, int NT>
__device__ inline void gemm_core(const float* __restrict__ A, int lda, int row0,
                                 const float* __restrict__ BT, int ldb, int col0,
                                 int K, v8f (&acc)[MT][NT], int lane) {
  for (int k0 = 0; k0 < K; k0 += 32) {
    v16bf af[MT], bfr[NT];
#pragma unroll
    for (int i = 0; i < MT; ++i) af[i] = frag_a_f32(A, lda, row0 + 16 * i, k0, lane);
#pragma unroll
    for (int j = 0; j < NT; ++j) bfr[j] = frag_b_f32(BT, ldb, col0 + 16 * j, k0, lane);
#pragma unroll
    for (int i = 0; i < MT; ++i)
#pragma unroll
      for (int j = 0; j < NT; ++j)
        acc[i][j] = WMMA_BF16(acc[i][j], af[i], bfr[j]);
  }
}

// ---- weight transpose prepass ----------------------------------------------
__global__ void transpose_f32(const float* __restrict__ in, float* __restrict__ out, int R, int C) {
  int idx = blockIdx.x * blockDim.x + threadIdx.x;
  if (idx >= R * C) return;
  int r = idx / C, c = idx % C;
  out[(size_t)c * R + r] = in[idx];
}

// ---- qkv = x @ Wqkv ; scatter to per-(b,h) Q(+bias), K, V^T ----------------
__global__ void qkv_gemm(const float* __restrict__ X, const float* __restrict__ WT,
                         const float* __restrict__ bias_pf,
                         float* __restrict__ Qb, float* __restrict__ Kh, float* __restrict__ VT) {
  const int lane = threadIdx.x & 31, wave = threadIdx.x >> 5;
  const int t = blockIdx.x * 8 + wave;
  const int ntc = QKVc / 64;                        // 24
  const int row0 = (t / ntc) * 32, col0 = (t % ntc) * 64;
  v8f acc[2][4] = {};
  gemm_core<2, 4>(X, Dc, row0, WT, Dc, col0, Dc, acc, lane);
  const int half = lane >> 4;
#pragma unroll
  for (int j = 0; j < 4; ++j) {
    const int gcol = col0 + 16 * j + (lane & 15);
    const int sect = gcol / HDc;                    // 0=q 1=k 2=v
    const int hd = gcol % HDc, h = hd >> 6, d = hd & 63;
#pragma unroll
    for (int i = 0; i < 2; ++i)
#pragma unroll
      for (int r = 0; r < 8; ++r) {
        const int grow = row0 + 16 * i + r + half * 8;
        const int b_ = grow >> 10, n_ = grow & (Nc - 1);
        const size_t bh = (size_t)(b_ * Hc + h);
        const float v = acc[i][j][r];
        if (sect == 0)      Qb[(bh * Nc + n_) * DHc + d] = v + bias_pf[hd];
        else if (sect == 1) Kh[(bh * Nc + n_) * DHc + d] = v;
        else                VT[(bh * DHc + d) * Nc + n_] = v;
      }
  }
}

// ---- Rh = r_t @ Wkr_t, scattered to [b,h,n,d] ------------------------------
__global__ void r_gemm(const float* __restrict__ Rt, const float* __restrict__ WT,
                       float* __restrict__ Rh) {
  const int lane = threadIdx.x & 31, wave = threadIdx.x >> 5;
  const int t = blockIdx.x * 8 + wave;
  const int ntc = HDc / 64;                         // 8
  const int row0 = (t / ntc) * 32, col0 = (t % ntc) * 64;
  v8f acc[2][4] = {};
  gemm_core<2, 4>(Rt, Dc, row0, WT, Dc, col0, Dc, acc, lane);
  const int half = lane >> 4;
#pragma unroll
  for (int j = 0; j < 4; ++j) {
    const int gcol = col0 + 16 * j + (lane & 15);
    const int h = gcol >> 6, d = gcol & 63;
#pragma unroll
    for (int i = 0; i < 2; ++i)
#pragma unroll
      for (int r = 0; r < 8; ++r) {
        const int grow = row0 + 16 * i + r + half * 8;
        const int b_ = grow >> 10, n_ = grow & (Nc - 1);
        Rh[(((size_t)(b_ * Hc + h)) * Nc + n_) * DHc + d] = acc[i][j][r];
      }
  }
}

// ---- QR chunk: rows [c0, c0+144) of Q·R^T per (b,h) ------------------------
__global__ void qr_chunk_gemm(const float* __restrict__ Qb, const float* __restrict__ Rh,
                              float* __restrict__ QRc, int c0) {
  const int lane = threadIdx.x & 31, wave = threadIdx.x >> 5;
  int t = blockIdx.x * 8 + wave;
  const int ntc = Nc / 64;                          // 16
  const int nt = t % ntc; t /= ntc;
  const int mt = t % (QR_ROWS / 16); t /= (QR_ROWS / 16);
  const int bh = t;
  const int prow0 = c0 + mt * 16;
  if (prow0 >= Nc) return;                          // wave-uniform
  const int col0 = nt * 64;
  const float* Q = Qb + (size_t)bh * Nc * DHc;
  const float* R = Rh + (size_t)bh * Nc * DHc;
  v8f acc[1][4] = {};
  gemm_core<1, 4>(Q, DHc, prow0, R, DHc, col0, DHc, acc, lane);
  const int half = lane >> 4;
#pragma unroll
  for (int j = 0; j < 4; ++j) {
    const int jc = col0 + 16 * j + (lane & 15);
#pragma unroll
    for (int r = 0; r < 8; ++r) {
      const int lp = mt * 16 + r + half * 8;
      QRc[((size_t)bh * QR_ROWS + lp) * Nc + jc] = acc[0][j][r];
    }
  }
}

// ---- scores: S = (Q·K^T + rel_shift(QR)) * SCALE, rows [c0,c0+128) ---------
__global__ void score_gemm(const float* __restrict__ Qb, const float* __restrict__ Kh,
                           const float* __restrict__ QRc, float* __restrict__ attn, int c0) {
  const int lane = threadIdx.x & 31, wave = threadIdx.x >> 5;
  int t = blockIdx.x * 8 + wave;
  const int ntc = Nc / 64;                          // 16
  const int nt = t % ntc; t /= ntc;
  const int mt = t % (CHUNK / 32); t /= (CHUNK / 32);
  const int bh = t;
  const int row0 = c0 + mt * 32, col0 = nt * 64;
  const float* Q  = Qb  + (size_t)bh * Nc * DHc;
  const float* Kp = Kh  + (size_t)bh * Nc * DHc;
  const float* QR = QRc + (size_t)bh * QR_ROWS * Nc;
  v8f acc[2][4] = {};
  gemm_core<2, 4>(Q, DHc, row0, Kp, DHc, col0, DHc, acc, lane);
  const int half = lane >> 4;
#pragma unroll
  for (int j = 0; j < 4; ++j) {
    const int q = col0 + 16 * j + (lane & 15);
#pragma unroll
    for (int i = 0; i < 2; ++i)
#pragma unroll
      for (int r = 0; r < 8; ++r) {
        const int p = row0 + 16 * i + r + half * 8;
        const int lp = p - c0;
        float bd;
        if (q <= p)          bd = QR[(size_t)lp * Nc + (q - p + Nc - 1)];
        else if (q == p + 1) bd = 0.0f;
        else                 bd = QR[(size_t)(lp + 1) * Nc + (q - p - 2)];
        attn[((size_t)bh * Nc + p) * Nc + q] = (acc[i][j][r] + bd) * SCALEc;
      }
  }
}

// ---- softmax over heads (axis=-1 of [i,j,b,h] in the reference) ------------
__global__ void softmax_h(float* __restrict__ attn) {
  const size_t idx = (size_t)blockIdx.x * blockDim.x + threadIdx.x;  // over B*N*N
  if (idx >= (size_t)Bc * Nc * Nc) return;
  const int j = (int)(idx & (Nc - 1));
  const int i = (int)((idx >> 10) & (Nc - 1));
  const int b = (int)(idx >> 20);
  const size_t base = (((size_t)b * Hc) * Nc + i) * Nc + j;
  const size_t str = (size_t)Nc * Nc;
  float s[Hc], m = -1e30f;
#pragma unroll
  for (int h = 0; h < Hc; ++h) { s[h] = attn[base + h * str]; m = fmaxf(m, s[h]); }
  float sum = 0.f;
#pragma unroll
  for (int h = 0; h < Hc; ++h) { s[h] = __expf(s[h] - m); sum += s[h]; }
  const float inv = 1.0f / sum;
#pragma unroll
  for (int h = 0; h < Hc; ++h) attn[base + h * str] = s[h] * inv;
}

// ---- O1[i, h*64+d] = sum_j P[i,j] V[j,d] per (b,h) -------------------------
__global__ void av_gemm(const float* __restrict__ attn, const float* __restrict__ VT,
                        float* __restrict__ O1) {
  const int lane = threadIdx.x & 31, wave = threadIdx.x >> 5;
  int t = blockIdx.x * 8 + wave;
  const int mt = t % (Nc / 32); t /= (Nc / 32);
  const int bh = t;
  const int row0 = mt * 32;
  const float* P  = attn + (size_t)bh * Nc * Nc;
  const float* Vt = VT   + (size_t)bh * DHc * Nc;
  v8f acc[2][4] = {};
  gemm_core<2, 4>(P, Nc, row0, Vt, Nc, 0, Nc, acc, lane);
  const int half = lane >> 4;
  const int b_ = bh >> 3, h = bh & 7;
#pragma unroll
  for (int j = 0; j < 4; ++j) {
    const int d = 16 * j + (lane & 15);
#pragma unroll
    for (int i = 0; i < 2; ++i)
#pragma unroll
      for (int r = 0; r < 8; ++r) {
        const int row = row0 + 16 * i + r + half * 8;
        O1[((size_t)(b_ * Nc + row)) * HDc + h * DHc + d] = acc[i][j][r];
      }
  }
}

// ---- out = O1 @ Wout + bout ------------------------------------------------
__global__ void out_gemm(const float* __restrict__ O1, const float* __restrict__ WoutT,
                         const float* __restrict__ bout, float* __restrict__ out) {
  const int lane = threadIdx.x & 31, wave = threadIdx.x >> 5;
  const int t = blockIdx.x * 8 + wave;
  const int ntc = Dc / 64;                          // 8
  const int row0 = (t / ntc) * 32, col0 = (t % ntc) * 64;
  v8f acc[2][4] = {};
  gemm_core<2, 4>(O1, HDc, row0, WoutT, HDc, col0, HDc, acc, lane);
  const int half = lane >> 4;
#pragma unroll
  for (int j = 0; j < 4; ++j) {
    const int col = col0 + 16 * j + (lane & 15);
    const float bb = bout[col];
#pragma unroll
    for (int i = 0; i < 2; ++i)
#pragma unroll
      for (int r = 0; r < 8; ++r) {
        const int row = row0 + 16 * i + r + half * 8;
        out[(size_t)row * Dc + col] = acc[i][j][r] + bb;
      }
  }
}

extern "C" void kernel_launch(void* const* d_in, const int* in_sizes, int n_in,
                              void* d_out, int out_size, void* d_ws, size_t ws_size,
                              hipStream_t stream) {
  const float* x       = (const float*)d_in[0];
  const float* r_t     = (const float*)d_in[1];
  /* d_in[2] r_p: dead in reference */
  const float* bias_pf = (const float*)d_in[3];
  const float* Wqkv    = (const float*)d_in[4];
  const float* Wkr_t   = (const float*)d_in[5];
  /* d_in[6] Wkr_p: dead in reference */
  const float* Wout    = (const float*)d_in[7];
  const float* bout    = (const float*)d_in[8];

  float* out  = (float*)d_out;                                // [B,N,D]
  float* attn = (float*)d_out + (size_t)Bc * Nc * Dc;         // [B,H,N,N]

  float* ws = (float*)d_ws;
  float* WqkvT = ws; ws += (size_t)QKVc * Dc;
  float* WkrT  = ws; ws += (size_t)HDc * Dc;
  float* WoutT = ws; ws += (size_t)Dc * HDc;
  float* Qb    = ws; ws += (size_t)Bc * Hc * Nc * DHc;
  float* Kh    = ws; ws += (size_t)Bc * Hc * Nc * DHc;
  float* VT    = ws; ws += (size_t)Bc * Hc * Nc * DHc;
  float* Rh    = ws; ws += (size_t)Bc * Hc * Nc * DHc;
  float* QRc   = ws; ws += (size_t)Bc * Hc * QR_ROWS * Nc;
  float* O1    = ws; ws += (size_t)Bc * Nc * HDc;             // total ~66 MB

  transpose_f32<<<(Dc * QKVc + 255) / 256, 256, 0, stream>>>(Wqkv, WqkvT, Dc, QKVc);
  transpose_f32<<<(Dc * HDc + 255) / 256, 256, 0, stream>>>(Wkr_t, WkrT, Dc, HDc);
  transpose_f32<<<(HDc * Dc + 255) / 256, 256, 0, stream>>>(Wout, WoutT, HDc, Dc);

  qkv_gemm<<<(Bc * Nc / 32) * (QKVc / 64) / 8, 256, 0, stream>>>(x, WqkvT, bias_pf, Qb, Kh, VT);
  r_gemm<<<(Bc * Nc / 32) * (HDc / 64) / 8, 256, 0, stream>>>(r_t, WkrT, Rh);

  for (int c0 = 0; c0 < Nc; c0 += CHUNK) {
    qr_chunk_gemm<<<(Bc * Hc) * (QR_ROWS / 16) * (Nc / 64) / 8, 256, 0, stream>>>(Qb, Rh, QRc, c0);
    score_gemm<<<(Bc * Hc) * (CHUNK / 32) * (Nc / 64) / 8, 256, 0, stream>>>(Qb, Kh, QRc, attn, c0);
  }

  softmax_h<<<(int)(((size_t)Bc * Nc * Nc + 255) / 256), 256, 0, stream>>>(attn);
  av_gemm<<<(Bc * Hc) * (Nc / 32) / 8, 256, 0, stream>>>(attn, VT, O1);
  out_gemm<<<(Bc * Nc / 32) * (Dc / 64) / 8, 256, 0, stream>>>(O1, WoutT, bout, out);
}
